// MultiHeadAttention_11012296147293
// MI455X (gfx1250) — matrix operand order repaired and checked by reading, not compiled, not run on hardware
//
#include <hip/hip_runtime.h>
#include <stdint.h>

// ---------------- problem constants ----------------
constexpr int BATCH = 4;
constexpr int SEQ   = 2048;
constexpr int DIM   = 1024;   // KDIM == NHEADS*VDIM == MDIM == 1024
constexpr int MT    = 32;     // query rows per workgroup
constexpr int NT    = 64;     // keys per inner block
constexpr int NBLK  = SEQ / NT;   // 32
constexpr int QBLKS = SEQ / MT;   // 64

// LDS row paddings (bank-conflict avoidance, keep 16B alignment)
constexpr int VT_P  = NT + 8;     // 72  -> 144B row stride (bank rotate 36)
constexpr int SC_P  = NT + 4;     // 68  -> 272B row stride (bank rotate 4)
constexpr int AT_P  = NT + 8;     // 72  -> 144B row stride
constexpr int CX_P  = DIM + 16;   // 1040 -> 2080B row stride (bank rotate 8)

typedef __bf16 bf16_t;
typedef __attribute__((ext_vector_type(16))) __bf16 v16bf;
typedef __attribute__((ext_vector_type(8)))  __bf16 v8bf;
typedef __attribute__((ext_vector_type(8)))  float  v8f;

union AFrag { v16bf v; v8bf h[2]; };
union BFrag { v16bf v; bf16_t e[16]; };
union Pack4 { bf16_t e[4]; unsigned long long u; };

// 8 consecutive f32 in global -> 8 bf16 (half of an A fragment)
__device__ __forceinline__ v8bf cvt8_g(const float* p) {
  const float4* p4 = (const float4*)p;
  float4 a = p4[0], b = p4[1];
  v8bf r;
  r[0]=(bf16_t)a.x; r[1]=(bf16_t)a.y; r[2]=(bf16_t)a.z; r[3]=(bf16_t)a.w;
  r[4]=(bf16_t)b.x; r[5]=(bf16_t)b.y; r[6]=(bf16_t)b.z; r[7]=(bf16_t)b.w;
  return r;
}

// 16 consecutive f32 in global -> 16 bf16 (full B fragment per lane)
__device__ __forceinline__ v16bf cvt16_g(const float* p) {
  const float4* p4 = (const float4*)p;
  BFrag r;
#pragma unroll
  for (int i = 0; i < 4; ++i) {
    float4 f = p4[i];
    r.e[4*i+0]=(bf16_t)f.x; r.e[4*i+1]=(bf16_t)f.y;
    r.e[4*i+2]=(bf16_t)f.z; r.e[4*i+3]=(bf16_t)f.w;
  }
  return r.v;
}

__global__ __launch_bounds__(256)
void mha_flash_wmma(const float* __restrict__ q,
                    const float* __restrict__ k,
                    const float* __restrict__ v,
                    const uint8_t* __restrict__ mask,
                    const float* __restrict__ W,     // (MDIM, DIM) row-major
                    float* __restrict__ out)
{
  // ---- LDS (~157 KB; 2 workgroups/WGP within CDNA5's 320 KB) ----
  __shared__ union {
    bf16_t Vt[DIM][VT_P];    // V^T tile, feature-major (key-loop phase) 144KB
    bf16_t ctx[MT][CX_P];    // ctx bf16 (out-projection phase)          65KB
  } big;
  __shared__ float  scoresLDS[MT][SC_P];
  __shared__ bf16_t attnLDS[MT][AT_P];
  __shared__ float  alphaLDS[MT];
  __shared__ float  lLDS[MT];

  const int tid   = threadIdx.x;
  const int wid   = tid >> 5;
  const int lane  = tid & 31;
  const int lhalf = lane >> 4;     // 0/1  (lane group)
  const int lmod  = lane & 15;

  const int b    = blockIdx.x / QBLKS;
  const int qblk = blockIdx.x % QBLKS;
  const int q0   = qblk * MT;

  const int r  = wid & 1;          // row group: rows [16r, 16r+16)
  const int cg = wid >> 1;         // col group 0..3

  const float scale = 0.03125f;    // 1/sqrt(1024)
  const float NEG_INF = -__builtin_inff();

  // online softmax state (owned by threads 0..31, one row each)
  float m_run = NEG_INF;
  float l_run = 0.0f;

  // ctx accumulators: rows 16r..16r+15, features cg*256..cg*256+255
  v8f ctx[16];
#pragma unroll
  for (int t = 0; t < 16; ++t) ctx[t] = (v8f){};

  const float* Qrow = q + ((size_t)b * SEQ + (q0 + r*16 + lmod)) * DIM;

  for (int jb = 0; jb < NBLK; ++jb) {
    const int k0 = jb * NT;

    // ---- cooperative stage: V block (64 keys x 1024 feats) -> LDS bf16, transposed.
    // keys interleaved across lanes; 4 keys packed per 8B store -> b64, conflict-free.
    {
      const int kq4 = (tid & 15) * 4;        // key quad base 0..60
      const int f0  = (tid >> 4) * 64;       // 16 feature groups of 64
      const float* vbase = v + ((size_t)b * SEQ + (k0 + kq4)) * DIM + f0;
      const float4* s0 = (const float4*)(vbase);
      const float4* s1 = (const float4*)(vbase + DIM);
      const float4* s2 = (const float4*)(vbase + 2*DIM);
      const float4* s3 = (const float4*)(vbase + 3*DIM);
#pragma unroll 4
      for (int fc = 0; fc < 16; ++fc) {
        float4 fa = s0[fc], fb = s1[fc], fg = s2[fc], fd = s3[fc];
        int feat = f0 + fc * 4;
        Pack4 p;
        p.e[0]=(bf16_t)fa.x; p.e[1]=(bf16_t)fb.x; p.e[2]=(bf16_t)fg.x; p.e[3]=(bf16_t)fd.x;
        *(unsigned long long*)&big.Vt[feat+0][kq4] = p.u;
        p.e[0]=(bf16_t)fa.y; p.e[1]=(bf16_t)fb.y; p.e[2]=(bf16_t)fg.y; p.e[3]=(bf16_t)fd.y;
        *(unsigned long long*)&big.Vt[feat+1][kq4] = p.u;
        p.e[0]=(bf16_t)fa.z; p.e[1]=(bf16_t)fb.z; p.e[2]=(bf16_t)fg.z; p.e[3]=(bf16_t)fd.z;
        *(unsigned long long*)&big.Vt[feat+2][kq4] = p.u;
        p.e[0]=(bf16_t)fa.w; p.e[1]=(bf16_t)fb.w; p.e[2]=(bf16_t)fg.w; p.e[3]=(bf16_t)fd.w;
        *(unsigned long long*)&big.Vt[feat+3][kq4] = p.u;
      }
    }

    // ---- scores: 16x16 subtile per wave, S = Q K^T ----
    {
      v8f sacc = (v8f){};
      const float* Krow = k + ((size_t)b * SEQ + (k0 + cg*16 + lmod)) * DIM + lhalf*16;
#pragma unroll 2
      for (int kc = 0; kc < DIM; kc += 32) {
        __builtin_prefetch((const void*)(Krow + kc + 64), 0, 1);
        AFrag a;
        const float* qp = Qrow + kc + lhalf*16;
        a.h[0] = cvt8_g(qp);
        a.h[1] = cvt8_g(qp + 8);
        v16bf bfrag = cvt16_g(Krow + kc);
        sacc = __builtin_amdgcn_wmma_f32_16x16x32_bf16(
                 false, a.v, false, bfrag, (short)0, sacc, false, false);
      }
      // apply scale + mask, store to LDS (row = i + 8*lhalf, col = lmod within subtile)
      const uint8_t* mp = mask + ((size_t)b * SEQ + (q0 + r*16 + 8*lhalf)) * SEQ
                               + (k0 + cg*16 + lmod);
#pragma unroll
      for (int i = 0; i < 8; ++i) {
        float sv = sacc[i] * scale;
        if (mp[(size_t)i * SEQ]) sv = NEG_INF;
        scoresLDS[r*16 + 8*lhalf + i][cg*16 + lmod] = sv;
      }
    }
    __syncthreads();

    // ---- online softmax over this key block (wave 0, one row per lane) ----
    if (tid < MT) {
      float bmax = NEG_INF;
      for (int j2 = 0; j2 < NT; ++j2) bmax = fmaxf(bmax, scoresLDS[tid][j2]);
      float mnew  = fmaxf(m_run, bmax);
      float alpha = __expf(m_run - mnew);
      float s = 0.0f;
      for (int j2 = 0; j2 < NT; ++j2) {
        float e = __expf(scoresLDS[tid][j2] - mnew);
        s += e;
        attnLDS[tid][j2] = (bf16_t)e;
      }
      l_run = l_run * alpha + s;
      m_run = mnew;
      alphaLDS[tid] = alpha;
    }
    __syncthreads();

    // ---- ctx: rescale then ctx += attn * V ----
    {
      float al[8];
#pragma unroll
      for (int i = 0; i < 8; ++i) al[i] = alphaLDS[r*16 + 8*lhalf + i];
#pragma unroll
      for (int t = 0; t < 16; ++t)
#pragma unroll
        for (int i = 0; i < 8; ++i) ctx[t][i] *= al[i];

#pragma unroll
      for (int kk = 0; kk < NT; kk += 32) {
        AFrag a;
        const bf16_t* ap = &attnLDS[r*16 + lmod][kk + lhalf*16];
        a.h[0] = *(const v8bf*)ap;
        a.h[1] = *(const v8bf*)(ap + 8);
#pragma unroll
        for (int t = 0; t < 16; ++t) {
          int feat = cg*256 + t*16 + lmod;
          AFrag bf;
          const bf16_t* bp = &big.Vt[feat][kk + lhalf*16];
          bf.h[0] = *(const v8bf*)bp;
          bf.h[1] = *(const v8bf*)(bp + 8);
          ctx[t] = __builtin_amdgcn_wmma_f32_16x16x32_bf16(
                     false, a.v, false, bf.v, (short)0, ctx[t], false, false);
        }
      }
    }
    __syncthreads();   // protect Vt / scores / attn before next block
  }

  // ---- finalize: ctx / l -> LDS bf16 (aliases Vt region) ----
  if (tid < MT) lLDS[tid] = l_run;
  __syncthreads();

  {
    float linv[8];
#pragma unroll
    for (int i = 0; i < 8; ++i) linv[i] = 1.0f / lLDS[r*16 + 8*lhalf + i];
#pragma unroll
    for (int t = 0; t < 16; ++t) {
      int feat = cg*256 + t*16 + lmod;
#pragma unroll
      for (int i = 0; i < 8; ++i)
        big.ctx[r*16 + 8*lhalf + i][feat] = (bf16_t)(ctx[t][i] * linv[i]);
    }
  }
  __syncthreads();

  // ---- out projection: O = ctx * W^T  (W is (out, in) row-major) ----
  // 8 col-groups of 128 outs; each wave computes BOTH 16-row tiles so every
  // W fragment is fetched exactly once per workgroup.
  v8f oacc[2][8];
#pragma unroll
  for (int rr = 0; rr < 2; ++rr)
#pragma unroll
    for (int t = 0; t < 8; ++t) oacc[rr][t] = (v8f){};

#pragma unroll 2
  for (int vc = 0; vc < DIM; vc += 32) {
    AFrag a0, a1;
    const bf16_t* ap0 = &big.ctx[lmod][vc + lhalf*16];
    a0.h[0] = *(const v8bf*)ap0;
    a0.h[1] = *(const v8bf*)(ap0 + 8);
    const bf16_t* ap1 = &big.ctx[16 + lmod][vc + lhalf*16];
    a1.h[0] = *(const v8bf*)ap1;
    a1.h[1] = *(const v8bf*)(ap1 + 8);
#pragma unroll
    for (int t = 0; t < 8; ++t) {
      int o = wid*128 + t*16 + lmod;
      const float* wp = W + (size_t)o * DIM + vc + lhalf*16;
      __builtin_prefetch((const void*)(wp + 64), 0, 1);
      v16bf bfrag = cvt16_g(wp);
      oacc[0][t] = __builtin_amdgcn_wmma_f32_16x16x32_bf16(
                     false, a0.v, false, bfrag, (short)0, oacc[0][t], false, false);
      oacc[1][t] = __builtin_amdgcn_wmma_f32_16x16x32_bf16(
                     false, a1.v, false, bfrag, (short)0, oacc[1][t], false, false);
    }
  }

#pragma unroll
  for (int rr = 0; rr < 2; ++rr)
#pragma unroll
    for (int t = 0; t < 8; ++t) {
      int o = wid*128 + t*16 + lmod;
#pragma unroll
      for (int i = 0; i < 8; ++i) {
        int row = q0 + rr*16 + 8*lhalf + i;
        out[((size_t)b * SEQ + row) * DIM + o] = oacc[rr][t][i];
      }
    }
}

extern "C" void kernel_launch(void* const* d_in, const int* in_sizes, int n_in,
                              void* d_out, int out_size, void* d_ws, size_t ws_size,
                              hipStream_t stream) {
  (void)in_sizes; (void)n_in; (void)d_ws; (void)ws_size; (void)out_size;
  const float*   q    = (const float*)d_in[0];
  const float*   k    = (const float*)d_in[1];
  const float*   v    = (const float*)d_in[2];
  const uint8_t* mask = (const uint8_t*)d_in[3];
  const float*   W    = (const float*)d_in[4];
  float* out = (float*)d_out;

  dim3 grid(BATCH * QBLKS);   // 256 workgroups
  dim3 block(256);            // 8 wave32s
  mha_flash_wmma<<<grid, block, 0, stream>>>(q, k, v, mask, W, out);
}
